// Block_48223892799907
// MI455X (gfx1250) — matrix operand order repaired and checked by reading, not compiled, not run on hardware
//
#include <hip/hip_runtime.h>
#include <stdint.h>

// ---------------------------------------------------------------------------
// Problem constants (match reference)
// ---------------------------------------------------------------------------
#define BB 2
#define TT 2048
#define CC 1024
#define HH 16
#define DD 64
#define EE 16
#define KK 2
#define NTOK (BB * TT)   // 4096

typedef __attribute__((ext_vector_type(16))) __bf16          v16bf;
typedef __attribute__((ext_vector_type(16))) unsigned short  v16u;
typedef __attribute__((ext_vector_type(8)))  unsigned short  v8u;
typedef __attribute__((ext_vector_type(8)))  unsigned int    v8ui;
typedef __attribute__((ext_vector_type(8)))  float           v8f;

// ---------------------------------------------------------------------------
// bf16 helpers: gfx1250 packed convert (1 VALU op for 2 converts)
// ---------------------------------------------------------------------------
__device__ __forceinline__ unsigned pack2(float a, float b) {
  unsigned r;
  asm("v_cvt_pk_bf16_f32 %0, %1, %2" : "=v"(r) : "v"(a), "v"(b));
  return r;   // low16 = bf16(a), high16 = bf16(b)
}
__device__ __forceinline__ unsigned short f32_bf16(float f) {
  unsigned r;
  asm("v_cvt_pk_bf16_f32 %0, %1, %2" : "=v"(r) : "v"(f), "v"(f));
  return (unsigned short)r;
}
__device__ __forceinline__ float bf16_f32(unsigned short h) {
  union { uint32_t u; float f; } x; x.u = ((uint32_t)h) << 16;
  return x.f;
}
// low 32 bits of a generic pointer to LDS == LDS byte offset (ISA 10.2)
__device__ __forceinline__ unsigned lds_off32(const void* p) {
  return (unsigned)(uintptr_t)p;
}

// ---------------------------------------------------------------------------
// WMMA wrapper: D = A(16x32 bf16) * B(32x16 bf16) + C(16x16 f32)
// ---------------------------------------------------------------------------
__device__ __forceinline__ v8f wmma_bf16(v16u a, v16u b, v8f c) {
  return __builtin_amdgcn_wmma_f32_16x16x32_bf16(
      false, __builtin_bit_cast(v16bf, a),
      false, __builtin_bit_cast(v16bf, b),
      (short)0, c, false, false);
}
// Build A fragment from contiguous-k LDS row: two 16B chunks
__device__ __forceinline__ v16u a_frag(const unsigned short* rowp, int hk) {
  v8u lo = *(const v8u*)(rowp + hk * 16);
  v8u hi = *(const v8u*)(rowp + hk * 16 + 8);
  return __builtin_shufflevector(lo, hi, 0,1,2,3,4,5,6,7,8,9,10,11,12,13,14,15);
}

// ---------------------------------------------------------------------------
// RMSNorm -> bf16
// ---------------------------------------------------------------------------
__global__ void __launch_bounds__(256)
rmsnorm_bf16_kernel(const float* __restrict__ x, const float* __restrict__ w,
                    unsigned short* __restrict__ out, int C_) {
  int row = blockIdx.x;
  const float* xr = x + (long)row * C_;
  __shared__ float red[8];
  float s = 0.f;
  for (int c = threadIdx.x; c < C_; c += 256) { float v = xr[c]; s += v * v; }
  for (int o = 16; o; o >>= 1) s += __shfl_xor(s, o, 32);
  if ((threadIdx.x & 31) == 0) red[threadIdx.x >> 5] = s;
  __syncthreads();
  if (threadIdx.x < 8) {
    float t = red[threadIdx.x];
    for (int o = 4; o; o >>= 1) t += __shfl_xor(t, o, 8);
    if (threadIdx.x == 0) red[0] = t;
  }
  __syncthreads();
  float inv = rsqrtf(red[0] / (float)C_ + 1e-5f);
  for (int c = threadIdx.x * 2; c < C_; c += 512) {
    unsigned* dst = (unsigned*)(out + (long)row * C_ + c);
    *dst = pack2(xr[c] * inv * w[c], xr[c + 1] * inv * w[c + 1]);
  }
}

__global__ void __launch_bounds__(256)
f32_to_bf16_kernel(const float* __restrict__ in, unsigned short* __restrict__ out, long n) {
  long i = ((long)blockIdx.x * 256 + threadIdx.x) * 2;
  if (i < n) *(unsigned*)(out + i) = pack2(in[i], in[i + 1]);
}

// ---------------------------------------------------------------------------
// Grouped GEMM, WMMA bf16. BM=64, BN=128, BK=32; 8 waves, each a 32x32 region
// (2x2 WMMA tiles). Full tiles: A staged by async-to-LDS DMA. W tile
// fp32->bf16 (v_cvt_pk_bf16_f32), transposed [n][k] -> B fragments are
// contiguous b128 LDS reads. k-loop specialized (async vs ragged) so the hot
// loop has no mode branches.
// ---------------------------------------------------------------------------
#define GBM 64
#define GBN 128
#define GBK 32

__global__ void __launch_bounds__(256)
gemm_bf16_kernel(const unsigned short* __restrict__ A,
                 const int* __restrict__ rowlist,   // null -> identity
                 const int* __restrict__ seg_cnt,   // null -> M from arg
                 const int* __restrict__ seg_off,
                 const float* __restrict__ W,
                 const float* __restrict__ bias,    // may be null
                 const float* __restrict__ residual,// may be null, [*,N]
                 float* __restrict__ outF,          // may be null
                 unsigned short* __restrict__ outB, // may be null
                 int M, int N, int K, int act)      // act: 0 none, 1 silu
{
  int z = blockIdx.z;
  int base = 0, Mz = M;
  if (seg_cnt) { Mz = seg_cnt[z]; base = seg_off[z]; }
  int m0 = blockIdx.x * GBM;
  if (m0 >= Mz) return;
  int n0 = blockIdx.y * GBN;
  long wz = (long)z * (long)K * (long)N;
  long bz = (long)z * (long)N;

  __shared__ unsigned short lA[GBM * GBK];    // [row][k]   4 KB
  __shared__ unsigned short lBt[GBN * GBK];   // [n][k]     8 KB (transposed)

  int tid = threadIdx.x;
  int lane = tid & 31;
  int wave = tid >> 5;
  int wm = wave >> 2;       // 0..1 : rows wm*32 + mt*16
  int wn = wave & 3;        // 0..3 : cols wn*32 + s*16
  int am = lane & 15;
  int hk = lane >> 4;

  v8f acc[2][2] = {{{}, {}}, {{}, {}}};
  bool full = (m0 + GBM) <= Mz;

  // ---- hoisted W-tile addressing (2 units/thread) ----
  int w_kp0 = tid >> 5,         w_nq0 = tid & 31;            // unit tid
  int w_kp1 = (tid + 256) >> 5, w_nq1 = tid & 31;            // unit tid+256
  const float* wp0 = W + wz + (long)(2 * w_kp0) * N + (n0 + 4 * w_nq0);
  const float* wp1 = W + wz + (long)(2 * w_kp1) * N + (n0 + 4 * w_nq1);
  unsigned* lbt32 = (unsigned*)lBt;
  const float* pfp = W + wz + n0 + (tid & 127) * 16;         // prefetch base

  auto stage_w = [&](int k0) {
    float4 a0 = *(const float4*)(wp0 + (long)k0 * N);
    float4 b0 = *(const float4*)(wp0 + (long)k0 * N + N);
    float4 a1 = *(const float4*)(wp1 + (long)k0 * N);
    float4 b1 = *(const float4*)(wp1 + (long)k0 * N + N);
    float av0[4] = {a0.x, a0.y, a0.z, a0.w}, bv0[4] = {b0.x, b0.y, b0.z, b0.w};
    float av1[4] = {a1.x, a1.y, a1.z, a1.w}, bv1[4] = {b1.x, b1.y, b1.z, b1.w};
#pragma unroll
    for (int j = 0; j < 4; ++j) {
      lbt32[(4 * w_nq0 + j) * (GBK / 2) + w_kp0] = pack2(av0[j], bv0[j]);
      lbt32[(4 * w_nq1 + j) * (GBK / 2) + w_kp1] = pack2(av1[j], bv1[j]);
    }
    if (k0 + GBK < K)
      __builtin_prefetch(pfp + (long)(k0 + GBK) * N, 0, 3);
  };
  auto tile_mma = [&]() {
    v16u afr[2], bfr[2];
#pragma unroll
    for (int mt = 0; mt < 2; ++mt)
      afr[mt] = a_frag(lA + (wm * 32 + mt * 16 + am) * GBK, hk);
#pragma unroll
    for (int s = 0; s < 2; ++s)
      bfr[s] = *(const v16u*)(lBt + (wn * 32 + s * 16 + am) * GBK + hk * 16);
#pragma unroll
    for (int mt = 0; mt < 2; ++mt)
#pragma unroll
      for (int s = 0; s < 2; ++s)
        acc[mt][s] = wmma_bf16(afr[mt], bfr[s], acc[mt][s]);
  };

  if (full) {
    // ---- async-DMA A path: hoisted addressing, branch-free hot loop ----
    int r = tid >> 2, c4 = tid & 3;                 // 64 rows x 4 x 16B
    int arow = rowlist ? rowlist[base + m0 + r] : (base + m0 + r);
    unsigned a_lds = lds_off32(&lA[r * GBK]) + c4 * 16;
    unsigned a_goff = (unsigned)((long)arow * K * 2) + c4 * 16;
    for (int k0 = 0; k0 < K; k0 += GBK) {
      unsigned goff = a_goff + (unsigned)(k0 * 2);
      asm volatile("global_load_async_to_lds_b128 %0, %1, %2"
                   :: "v"(a_lds), "v"(goff), "s"(A) : "memory");
      stage_w(k0);
      asm volatile("s_wait_asynccnt 0" ::: "memory");
      __syncthreads();
      tile_mma();
      __syncthreads();
    }
  } else {
    // ---- ragged tail path: zero-filled scalar staging (2 units/thread) ----
    int f_r0 = tid >> 3, f_c = (tid & 7) * 4;
    int f_r1 = (tid + 256) >> 3;
    long f_a0 = -1, f_a1 = -1;
    if (m0 + f_r0 < Mz) f_a0 = rowlist ? rowlist[base + m0 + f_r0] : (base + m0 + f_r0);
    if (m0 + f_r1 < Mz) f_a1 = rowlist ? rowlist[base + m0 + f_r1] : (base + m0 + f_r1);
    for (int k0 = 0; k0 < K; k0 += GBK) {
      ushort4 v0 = {0, 0, 0, 0}, v1 = {0, 0, 0, 0};
      if (f_a0 >= 0) v0 = *(const ushort4*)(A + f_a0 * K + k0 + f_c);
      if (f_a1 >= 0) v1 = *(const ushort4*)(A + f_a1 * K + k0 + f_c);
      *(ushort4*)(lA + f_r0 * GBK + f_c) = v0;
      *(ushort4*)(lA + f_r1 * GBK + f_c) = v1;
      stage_w(k0);
      __syncthreads();
      tile_mma();
      __syncthreads();
    }
  }

  // ---- epilogue ----
#pragma unroll
  for (int mt = 0; mt < 2; ++mt) {
#pragma unroll
    for (int r = 0; r < 8; ++r) {
      int row = m0 + wm * 32 + mt * 16 + r + 8 * hk;
      if (row >= Mz) continue;
      long orow = (long)(base + row);
#pragma unroll
      for (int s = 0; s < 2; ++s) {
        int col = n0 + wn * 32 + s * 16 + am;
        float v = acc[mt][s][r];
        if (bias) v += bias[bz + col];
        if (act == 1) v = v / (1.f + __expf(-v));     // silu
        if (residual) v += residual[orow * N + col];
        if (outF) outF[orow * N + col] = v;
        if (outB) outB[orow * N + col] = f32_bf16(v);
      }
    }
  }
}

// ---------------------------------------------------------------------------
// RoPE on q,k halves of qkv (in place)
// ---------------------------------------------------------------------------
__global__ void __launch_bounds__(256)
rope_kernel(float* __restrict__ qkv) {
  long i = (long)blockIdx.x * 256 + threadIdx.x;
  long total = (long)NTOK * HH * (DD / 2);
  if (i >= total) return;
  int pair = (int)(i % (DD / 2));
  long t2 = i / (DD / 2);
  int h = (int)(t2 % HH);
  long n = t2 / HH;
  int pos = (int)(n % TT);
  float freq = __powf(10000.f, -2.f * (float)pair / (float)DD);
  float sn, cs;
  __sincosf((float)pos * freq, &sn, &cs);
  long baseq = n * (3l * CC) + h * DD + 2 * pair;
  long basek = baseq + CC;
  float q0 = qkv[baseq], q1 = qkv[baseq + 1];
  qkv[baseq]     = q0 * cs - q1 * sn;
  qkv[baseq + 1] = q0 * sn + q1 * cs;
  float k0 = qkv[basek], k1 = qkv[basek + 1];
  qkv[basek]     = k0 * cs - k1 * sn;
  qkv[basek + 1] = k0 * sn + k1 * cs;
}

// ---------------------------------------------------------------------------
// Flash attention: 8 waves x 16 queries per block; 32-key chunks staged in LDS
// as bf16 (K as [key][d], V transposed [d][key]); QK^T and PV via WMMA.
// ---------------------------------------------------------------------------
#define AQ 16
#define AW 8
#define ACH 32

__global__ void __launch_bounds__(256)
attn_kernel(const float* __restrict__ qkv, float* __restrict__ y) {
  int b = blockIdx.z;
  int h = blockIdx.y;
  int q0blk = blockIdx.x * (AQ * AW);
  int tid = threadIdx.x, lane = tid & 31, wave = tid >> 5;
  int qw = q0blk + wave * AQ;

  __shared__ unsigned short lK[ACH * 64];    // [key][d]   4 KB
  __shared__ unsigned short lVt[64 * ACH];   // [d][key]   4 KB
  __shared__ unsigned short lP[AW][AQ * ACH];// per-wave P [row][key] 8 KB

  const float scale = 0.125f;                        // 1/sqrt(64)
  long rowstride = 3l * CC;
  long qbase = (long)b * TT * rowstride;
  int am = lane & 15;
  int hk = lane >> 4;

  // Q fragments (16x64 -> two 16x32 A fragments), scaled, packed converts
  v16u qa0, qa1;
  {
    long qrow = qbase + (long)(qw + am) * rowstride + h * DD;
    const float* qp = qkv + qrow;
    v8ui q0p, q1p;
#pragma unroll
    for (int half = 0; half < 2; ++half) {     // k chunks [hk*16,+8) and [hk*16+8,+8)
      float4 f0 = *(const float4*)(qp + half * 8 + hk * 16);
      float4 f1 = *(const float4*)(qp + half * 8 + hk * 16 + 4);
      float4 g0 = *(const float4*)(qp + 32 + half * 8 + hk * 16);
      float4 g1 = *(const float4*)(qp + 32 + half * 8 + hk * 16 + 4);
      q0p[half * 4 + 0] = pack2(f0.x * scale, f0.y * scale);
      q0p[half * 4 + 1] = pack2(f0.z * scale, f0.w * scale);
      q0p[half * 4 + 2] = pack2(f1.x * scale, f1.y * scale);
      q0p[half * 4 + 3] = pack2(f1.z * scale, f1.w * scale);
      q1p[half * 4 + 0] = pack2(g0.x * scale, g0.y * scale);
      q1p[half * 4 + 1] = pack2(g0.z * scale, g0.w * scale);
      q1p[half * 4 + 2] = pack2(g1.x * scale, g1.y * scale);
      q1p[half * 4 + 3] = pack2(g1.z * scale, g1.w * scale);
    }
    qa0 = __builtin_bit_cast(v16u, q0p);
    qa1 = __builtin_bit_cast(v16u, q1p);
  }

  v8f o0 = {}, o1 = {}, o2 = {}, o3 = {};
  float m_run[8], l_run[8];
#pragma unroll
  for (int r = 0; r < 8; ++r) { m_run[r] = -1e30f; l_run[r] = 0.f; }

  int nChunks = (q0blk + AQ * AW) / ACH;             // causal bound for block
  for (int ch = 0; ch < nChunks; ++ch) {
    int kstart = ch * ACH;
    // ---- stage K ([key][d]) with packed pair stores ----
    for (int i = tid; i < ACH * 32; i += 256) {      // 1024 units
      int kk = i >> 5, dp = i & 31;
      long krow = qbase + (long)(kstart + kk) * rowstride + h * DD;
      float2 kv = *(const float2*)(qkv + krow + CC + 2 * dp);
      ((unsigned*)lK)[kk * 32 + dp] = pack2(kv.x, kv.y);
    }
    // ---- stage V transposed ([d][key]) ----
    {
      int kp = tid >> 4;                // 0..15 key pair
      int dq = tid & 15;                // 0..15 d quad
      long krow = qbase + (long)(kstart + 2 * kp) * rowstride + h * DD + 2l * CC;
      float4 a = *(const float4*)(qkv + krow + 4 * dq);
      float4 bq = *(const float4*)(qkv + krow + rowstride + 4 * dq);
      float av[4] = {a.x, a.y, a.z, a.w};
      float bv[4] = {bq.x, bq.y, bq.z, bq.w};
#pragma unroll
      for (int j = 0; j < 4; ++j)
        ((unsigned*)lVt)[(4 * dq + j) * (ACH / 2) + kp] = pack2(av[j], bv[j]);
    }
    __syncthreads();

    if (kstart <= qw + AQ - 1) {
      // S tiles: keys 0..15 and 16..31; inner dim d=64 -> 2 WMMA steps each
      v8f s0 = {}, s1 = {};
#pragma unroll
      for (int s = 0; s < 2; ++s) {
        v16u bk0 = *(const v16u*)(lK + am * 64 + s * 32 + hk * 16);
        v16u bk1 = *(const v16u*)(lK + (16 + am) * 64 + s * 32 + hk * 16);
        s0 = wmma_bf16(s ? qa1 : qa0, bk0, s0);
        s1 = wmma_bf16(s ? qa1 : qa0, bk1, s1);
      }
      // causal mask + online softmax (rows striped: M = r + 8*hk)
#pragma unroll
      for (int r = 0; r < 8; ++r) {
        int qrow = qw + r + 8 * hk;
        int key0 = kstart + am;
        float v0 = s0[r]; if (key0 > qrow) v0 = -1e30f;
        float v1 = s1[r]; if (key0 + 16 > qrow) v1 = -1e30f;
        float mrow = fmaxf(v0, v1);
        for (int o = 1; o < 16; o <<= 1) mrow = fmaxf(mrow, __shfl_xor(mrow, o, 32));
        float mnew = fmaxf(m_run[r], mrow);
        float alpha = __expf(m_run[r] - mnew);
        m_run[r] = mnew;
        l_run[r] *= alpha;
        o0[r] *= alpha; o1[r] *= alpha; o2[r] *= alpha; o3[r] *= alpha;
        float p0 = __expf(v0 - mnew);
        float p1 = __expf(v1 - mnew);
        float rs = p0 + p1;
        for (int o = 1; o < 16; o <<= 1) rs += __shfl_xor(rs, o, 32);
        l_run[r] += rs;
        lP[wave][(r + 8 * hk) * ACH + am]      = f32_bf16(p0);
        lP[wave][(r + 8 * hk) * ACH + 16 + am] = f32_bf16(p1);
      }
      // wave-private LDS RAW: order + drain DS before re-reading
      __builtin_amdgcn_wave_barrier();
      asm volatile("s_wait_dscnt 0" ::: "memory");
      __builtin_amdgcn_wave_barrier();

      v16u pf = a_frag(&lP[wave][am * ACH], hk);
#pragma unroll
      for (int nt = 0; nt < 4; ++nt) {
        v16u vf = *(const v16u*)(lVt + (nt * 16 + am) * ACH + hk * 16);
        v8f& oacc = (nt == 0) ? o0 : (nt == 1) ? o1 : (nt == 2) ? o2 : o3;
        oacc = wmma_bf16(pf, vf, oacc);
      }
    }
    __syncthreads();
  }

  {
#pragma unroll
    for (int r = 0; r < 8; ++r) {
      int qrow = qw + r + 8 * hk;
      float inv = 1.f / l_run[r];
      long orow = ((long)b * TT + qrow) * (long)CC + h * DD;
      y[orow + 0 * 16 + am] = o0[r] * inv;
      y[orow + 1 * 16 + am] = o1[r] * inv;
      y[orow + 2 * 16 + am] = o2[r] * inv;
      y[orow + 3 * 16 + am] = o3[r] * inv;
    }
  }
}

// ---------------------------------------------------------------------------
// MoE gating: softmax over 16 logits, top-2 by (p + gate_bias), weight = p.
// ---------------------------------------------------------------------------
__global__ void __launch_bounds__(256)
gate_kernel(const unsigned short* __restrict__ h2, const float* __restrict__ gate_w,
            const float* __restrict__ gate_bias,
            int* __restrict__ topk_idx, float* __restrict__ topk_w,
            int* __restrict__ posbuf, int* __restrict__ counts,
            float* __restrict__ psum) {
  int token = blockIdx.x * 8 + (threadIdx.x >> 5);
  int lane = threadIdx.x & 31;
  if (token >= NTOK) return;
  float logit[EE];
  const unsigned short* hr = h2 + (long)token * CC;
#pragma unroll
  for (int e = 0; e < EE; ++e) {
    float s = 0.f;
    const float* wr = gate_w + e * CC;
    for (int c = lane; c < CC; c += 32) s += bf16_f32(hr[c]) * wr[c];
    for (int o = 16; o; o >>= 1) s += __shfl_xor(s, o, 32);
    logit[e] = s;
  }
  if (lane == 0) {
    float mx = logit[0];
    for (int e = 1; e < EE; ++e) mx = fmaxf(mx, logit[e]);
    float p[EE], sum = 0.f;
    for (int e = 0; e < EE; ++e) { p[e] = __expf(logit[e] - mx); sum += p[e]; }
    float inv = 1.f / sum;
    for (int e = 0; e < EE; ++e) { p[e] *= inv; atomicAdd(&psum[e], p[e]); }
    int i0 = 0; float b0v = p[0] + gate_bias[0];
    for (int e = 1; e < EE; ++e) {
      float v = p[e] + gate_bias[e];
      if (v > b0v) { b0v = v; i0 = e; }
    }
    int i1 = -1; float b1v = -1e30f;
    for (int e = 0; e < EE; ++e) {
      if (e == i0) continue;
      float v = p[e] + gate_bias[e];
      if (v > b1v) { b1v = v; i1 = e; }
    }
    int idx2[2] = { i0, i1 };
    for (int k = 0; k < 2; ++k) {
      int e = idx2[k];
      topk_idx[token * 2 + k] = e;
      topk_w[token * 2 + k] = p[e];
      posbuf[token * 2 + k] = atomicAdd(&counts[e], 1);
    }
  }
}

__global__ void zero16_kernel(int* counts, float* psum) {
  int i = threadIdx.x;
  if (i < EE) { counts[i] = 0; psum[i] = 0.f; }
}

__global__ void prefix_kernel(const int* counts, int* offsets) {
  if (threadIdx.x == 0) {
    int acc = 0;
    for (int e = 0; e < EE; ++e) { offsets[e] = acc; acc += counts[e]; }
    offsets[EE] = acc;
  }
}

__global__ void __launch_bounds__(256)
fill_rows_kernel(const int* __restrict__ topk_idx, const int* __restrict__ posbuf,
                 const int* __restrict__ offsets, int* __restrict__ rowlist,
                 int* __restrict__ slotrow) {
  int i = blockIdx.x * 256 + threadIdx.x;
  if (i >= NTOK * KK) return;
  int e = topk_idx[i];
  int row = offsets[e] + posbuf[i];
  rowlist[row] = i >> 1;      // token id
  slotrow[i] = row;
}

// ---------------------------------------------------------------------------
// Final combine: out = x2 + shared_mlp + sum_k w_k * expert_out[slot]
// ---------------------------------------------------------------------------
__global__ void __launch_bounds__(256)
finalize_kernel(const float* __restrict__ x2, const float* __restrict__ sh_out,
                const float* __restrict__ moe_out, const int* __restrict__ slotrow,
                const float* __restrict__ topk_w, float* __restrict__ out) {
  long i = (long)blockIdx.x * 256 + threadIdx.x;
  long total = (long)NTOK * CC;
  if (i >= total) return;
  long t = i / CC;
  int c = (int)(i % CC);
  float v = x2[i] + sh_out[i];
#pragma unroll
  for (int k = 0; k < KK; ++k) {
    long r = slotrow[t * KK + k];
    v += topk_w[t * KK + k] * moe_out[r * CC + c];
  }
  out[i] = v;
}

__global__ void loss_kernel(const int* counts, const float* psum, float* out) {
  if (threadIdx.x == 0) {
    float loss = 0.f;
    for (int e = 0; e < EE; ++e) {
      float f = (float)counts[e] * (float)EE / ((float)KK * NTOK + 1e-6f);
      loss += f * (psum[e] / (float)NTOK);
    }
    out[(long)NTOK * CC] = 0.01f * loss;
  }
}

// ---------------------------------------------------------------------------
// Host orchestration
// ---------------------------------------------------------------------------
extern "C" void kernel_launch(void* const* d_in, const int* in_sizes, int n_in,
                              void* d_out, int out_size, void* d_ws, size_t ws_size,
                              hipStream_t stream) {
  const float* x       = (const float*)d_in[0];
  const float* ln1_w   = (const float*)d_in[1];
  const float* ln2_w   = (const float*)d_in[2];
  const float* w_attn  = (const float*)d_in[3];
  const float* b_attn  = (const float*)d_in[4];
  const float* w_proj  = (const float*)d_in[5];
  const float* b_proj  = (const float*)d_in[6];
  const float* gate_w  = (const float*)d_in[7];
  const float* gate_b  = (const float*)d_in[8];
  const float* exp_wf  = (const float*)d_in[9];
  const float* exp_bf  = (const float*)d_in[10];
  const float* exp_wp  = (const float*)d_in[11];
  const float* exp_bp  = (const float*)d_in[12];
  const float* sh_wf   = (const float*)d_in[13];
  const float* sh_bf   = (const float*)d_in[14];
  const float* sh_wp   = (const float*)d_in[15];
  const float* sh_bp   = (const float*)d_in[16];
  float* out = (float*)d_out;

  char* ws = (char*)d_ws;
  size_t off = 0;
  auto alloc = [&](size_t bytes) {
    size_t o = off;
    off = (off + bytes + 255) & ~(size_t)255;
    return o;
  };
  const long N = NTOK;
  unsigned short* h1   = (unsigned short*)(ws + alloc(N * CC * 2));
  float*          qkv  = (float*)         (ws + alloc(N * 3l * CC * 4));
  float*          y    = (float*)         (ws + alloc(N * CC * 4));
  unsigned short* ybf  = (unsigned short*)(ws + alloc(N * CC * 2));
  float*          x2   = (float*)         (ws + alloc(N * CC * 4));
  unsigned short* h2   = (unsigned short*)(ws + alloc(N * CC * 2));
  unsigned short* tbuf = (unsigned short*)(ws + alloc(N * KK * 2l * CC * 2));
  float*          moeo = (float*)         (ws + alloc(N * KK * (long)CC * 4));
  unsigned short* sht  = (unsigned short*)(ws + alloc(N * 2l * CC * 2));
  float*          sho  = (float*)         (ws + alloc(N * CC * 4));
  int*            counts  = (int*)  (ws + alloc((EE) * 4));
  int*            offsets = (int*)  (ws + alloc((EE + 1) * 4));
  float*          psum    = (float*)(ws + alloc(EE * 4));
  int*            tkidx   = (int*)  (ws + alloc(N * KK * 4));
  float*          tkw     = (float*)(ws + alloc(N * KK * 4));
  int*            posbuf  = (int*)  (ws + alloc(N * KK * 4));
  int*            rowlist = (int*)  (ws + alloc(N * KK * 4));
  int*            slotrow = (int*)  (ws + alloc(N * KK * 4));
  (void)ws_size; (void)n_in; (void)in_sizes; (void)out_size;

  // 1) h1 = rmsnorm(x, ln1_w) -> bf16
  rmsnorm_bf16_kernel<<<N, 256, 0, stream>>>(x, ln1_w, h1, CC);

  // 2) qkv = h1 @ w_attn + b_attn   (M=4096, K=1024, N=3072)
  gemm_bf16_kernel<<<dim3(N / GBM, 3 * CC / GBN, 1), 256, 0, stream>>>(
      h1, nullptr, nullptr, nullptr, w_attn, b_attn, nullptr,
      qkv, nullptr, (int)N, 3 * CC, CC, 0);

  // 3) RoPE on q,k in place
  {
    long total = N * HH * (DD / 2);
    rope_kernel<<<(int)((total + 255) / 256), 256, 0, stream>>>(qkv);
  }

  // 4) flash attention -> y
  attn_kernel<<<dim3(TT / (AQ * AW), HH, BB), 256, 0, stream>>>(qkv, y);

  // 5) y -> bf16
  f32_to_bf16_kernel<<<(int)((N * CC / 2 + 255) / 256), 256, 0, stream>>>(y, ybf, N * CC);

  // 6) x2 = x + y @ w_proj + b_proj   (M=4096, K=1024, N=1024)
  gemm_bf16_kernel<<<dim3(N / GBM, CC / GBN, 1), 256, 0, stream>>>(
      ybf, nullptr, nullptr, nullptr, w_proj, b_proj, x,
      x2, nullptr, (int)N, CC, CC, 0);

  // 7) h2 = rmsnorm(x2, ln2_w) -> bf16
  rmsnorm_bf16_kernel<<<N, 256, 0, stream>>>(x2, ln2_w, h2, CC);

  // 8) gating + routing compaction
  zero16_kernel<<<1, 32, 0, stream>>>(counts, psum);
  gate_kernel<<<(int)(N / 8), 256, 0, stream>>>(h2, gate_w, gate_b,
                                                tkidx, tkw, posbuf, counts, psum);
  prefix_kernel<<<1, 32, 0, stream>>>(counts, offsets);
  fill_rows_kernel<<<(int)(N * KK / 256), 256, 0, stream>>>(tkidx, posbuf, offsets,
                                                            rowlist, slotrow);

  // 9) expert MLPs (grouped, ragged M via counts): t = silu(h2_g @ wf + bf) -> bf16
  gemm_bf16_kernel<<<dim3(N / GBM, 2 * CC / GBN, EE), 256, 0, stream>>>(
      h2, rowlist, counts, offsets, exp_wf, exp_bf, nullptr,
      nullptr, tbuf, (int)N, 2 * CC, CC, 1);
  //    moeo = t @ wp + bp
  gemm_bf16_kernel<<<dim3(N / GBM, CC / GBN, EE), 256, 0, stream>>>(
      tbuf, nullptr, counts, offsets, exp_wp, exp_bp, nullptr,
      moeo, nullptr, (int)N, CC, 2 * CC, 0);

  // 10) shared expert: sht = silu(h2 @ sh_wf + sh_bf); sho = sht @ sh_wp + sh_bp
  gemm_bf16_kernel<<<dim3(N / GBM, 2 * CC / GBN, 1), 256, 0, stream>>>(
      h2, nullptr, nullptr, nullptr, sh_wf, sh_bf, nullptr,
      nullptr, sht, (int)N, 2 * CC, CC, 1);
  gemm_bf16_kernel<<<dim3(N / GBM, CC / GBN, 1), 256, 0, stream>>>(
      sht, nullptr, nullptr, nullptr, sh_wp, sh_bp, nullptr,
      sho, nullptr, (int)N, CC, 2 * CC, 0);

  // 11) combine + aux loss
  finalize_kernel<<<(int)((N * CC + 255) / 256), 256, 0, stream>>>(
      x2, sho, moeo, slotrow, tkw, out);
  loss_kernel<<<1, 32, 0, stream>>>(counts, psum, out);
}
